// DeepSeekMoE_64278480552165
// MI455X (gfx1250) — matrix operand order repaired and checked by reading, not compiled, not run on hardware
//
#include <hip/hip_runtime.h>
#include <hip/hip_bf16.h>
#include <stdint.h>

// ---------------------------------------------------------------------------
// DeepSeek-MoE block for gfx1250 (MI455X).
//  - Router: softmax -> top-6 -> renormalize -> per-expert token lists (atomic
//    scatter; output is order-invariant, hence deterministic).
//  - Shared experts: dense WMMA GEMMs (scale 1/NUM_SHARED).
//  - Routed experts: gathered WMMA GEMMs over each expert's token list only.
//  - GEMM: 128x128x32 block tiles double-buffered in LDS, filled with
//    GLOBAL_LOAD_ASYNC_TO_LDS_B128 (fallback: ds_store staging), consumed by
//    v_wmma_f32_16x16x32_f16 (8 per wave per k-step).
// ---------------------------------------------------------------------------

#define D_MODEL    2048
#define D_HIDDEN   1408
#define NUM_SHARED 2
#define NUM_ROUTED 32
#define TOP_K      6
#define T_TOKENS   4096   // 2 * 2048

#if defined(__has_builtin)
#if __has_builtin(__builtin_amdgcn_global_load_async_to_lds_b128)
#define MOE_HAS_ASYNC 1
#endif
#endif
#ifndef MOE_HAS_ASYNC
#define MOE_HAS_ASYNC 0
#endif

typedef __attribute__((ext_vector_type(16))) _Float16 v16h;
typedef __attribute__((ext_vector_type(8)))  _Float16 v8h;
typedef __attribute__((ext_vector_type(8)))  float    v8f;
typedef __attribute__((ext_vector_type(4)))  int      v4i;

typedef __attribute__((address_space(1))) v4i as1_v4i;
typedef __attribute__((address_space(3))) v4i as3_v4i;

union HFrag { v16h v; v8h h[2]; };

// LDS row stride in halfs: 32 data halfs + 8 pad = 80B rows (16B aligned, bank spread)
#define RS 40

// Stage one 16-byte chunk global -> LDS.
__device__ inline void stage16(const _Float16* __restrict__ g, _Float16* l) {
#if MOE_HAS_ASYNC
  __builtin_amdgcn_global_load_async_to_lds_b128((as1_v4i*)g, (as3_v4i*)l, 0, 0);
#else
  *(v8h*)l = *(const v8h*)g;
#endif
}

__device__ inline void wait_async_all() {
#if MOE_HAS_ASYNC
#if __has_builtin(__builtin_amdgcn_s_wait_asynccnt)
  __builtin_amdgcn_s_wait_asynccnt(0);
#else
  asm volatile("s_wait_asynccnt 0x0" ::: "memory");
#endif
#endif
}

// ---------------------------------------------------------------------------
// Utility kernels
// ---------------------------------------------------------------------------
__global__ void moe_zero_b32(unsigned* __restrict__ p, size_t n) {
  size_t i = (size_t)blockIdx.x * blockDim.x + threadIdx.x;
  if (i < n) p[i] = 0u;
}

__global__ void moe_f32_to_f16(const float* __restrict__ src,
                               _Float16* __restrict__ dst, size_t n) {
  size_t i = (size_t)blockIdx.x * blockDim.x + threadIdx.x;
  if (i < n) dst[i] = (_Float16)src[i];
}

// Transpose-convert: src fp32 [R][C] row-major -> dst f16 [C][R] row-major.
__global__ void moe_transpose_f16(const float* __restrict__ src,
                                  _Float16* __restrict__ dst, int R, int C) {
  __shared__ float tile[32][33];
  const int tx = threadIdx.x, ty = threadIdx.y;
  const int c0 = blockIdx.x * 32, r0 = blockIdx.y * 32;
#pragma unroll
  for (int i = 0; i < 4; ++i)
    tile[ty + i * 8][tx] = src[(size_t)(r0 + ty + i * 8) * C + (c0 + tx)];
  __syncthreads();
#pragma unroll
  for (int i = 0; i < 4; ++i)
    dst[(size_t)(c0 + ty + i * 8) * R + (r0 + tx)] = (_Float16)tile[tx][ty + i * 8];
}

// ---------------------------------------------------------------------------
// Router: logits -> softmax -> top-6 -> renormalize; scatter (token, weight)
// into per-expert lists.  One token per thread, all state in registers.
// ---------------------------------------------------------------------------
__global__ __launch_bounds__(256) void moe_router_scatter(
    const float* __restrict__ x, const float* __restrict__ rw,
    const float* __restrict__ rb,
    int* __restrict__ cnt,        // [NUM_ROUTED]
    int* __restrict__ list,       // [NUM_ROUTED][T]
    float* __restrict__ gval) {   // [NUM_ROUTED][T]
  const int t = blockIdx.x * 256 + threadIdx.x;
  if (t >= T_TOKENS) return;

  float acc[NUM_ROUTED];
#pragma unroll
  for (int e = 0; e < NUM_ROUTED; ++e) acc[e] = rb[e];

  const float* xr = x + (size_t)t * D_MODEL;
  for (int k = 0; k < D_MODEL; ++k) {
    const float xv = xr[k];
    const float* wr = rw + (size_t)k * NUM_ROUTED;
#pragma unroll
    for (int e = 0; e < NUM_ROUTED; ++e) acc[e] = fmaf(xv, wr[e], acc[e]);
  }

  float mx = acc[0];
#pragma unroll
  for (int e = 1; e < NUM_ROUTED; ++e) mx = fmaxf(mx, acc[e]);
  float sum = 0.0f;
#pragma unroll
  for (int e = 0; e < NUM_ROUTED; ++e) { acc[e] = __expf(acc[e] - mx); sum += acc[e]; }
  const float inv = 1.0f / sum;
#pragma unroll
  for (int e = 0; e < NUM_ROUTED; ++e) acc[e] *= inv;

  unsigned used = 0u;
  float tv[TOP_K]; int ti[TOP_K];
  float tsum = 0.0f;
#pragma unroll
  for (int s = 0; s < TOP_K; ++s) {
    float best = -1.0f; int bi = 0;
#pragma unroll
    for (int e = 0; e < NUM_ROUTED; ++e) {
      const bool avail = ((used >> e) & 1u) == 0u;
      if (avail && acc[e] > best) { best = acc[e]; bi = e; }
    }
    used |= (1u << bi);
    tv[s] = best; ti[s] = bi; tsum += best;
  }
  const float rinv = 1.0f / tsum;
#pragma unroll
  for (int s = 0; s < TOP_K; ++s) {
    const int e = ti[s];
    const int pos = atomicAdd(&cnt[e], 1);
    list[(size_t)e * T_TOKENS + pos] = t;
    gval[(size_t)e * T_TOKENS + pos] = tv[s] * rinv;
  }
}

// ---------------------------------------------------------------------------
// WMMA GEMM #1 (gathered): Hout[p][n] = relu(sum_k A[tok(p)][k]*Bt[n][k]+b[n])
// rowList==null -> identity gather; cntPtr==null -> cnt = Mtotal.
// Fixed grid; blocks beyond the (device-side) row count exit early.
// ---------------------------------------------------------------------------
__global__ __launch_bounds__(256) void moe_gemm1_relu(
    const _Float16* __restrict__ A, const _Float16* __restrict__ Bt,
    const float* __restrict__ bias, _Float16* __restrict__ Hout,
    const int* __restrict__ rowList, const int* __restrict__ cntPtr,
    int Mtotal, int N, int K) {
  const int cnt = cntPtr ? *cntPtr : Mtotal;
  const int mblk = blockIdx.y * 128;
  if (mblk >= cnt) return;   // uniform over the block (EXEC stays full)

  __shared__ __align__(16) _Float16 sA[2][128 * RS];
  __shared__ __align__(16) _Float16 sB[2][128 * RS];

  const int tid = threadIdx.x;
  const int lane = tid & 31, wave = tid >> 5;
  const int wm = wave >> 1, wn = wave & 1;
  const int lm = lane & 15, kh = lane >> 4;
  const int nblk = blockIdx.x * 128;

  // staging: thread -> (row, 32B half-row); 128 rows x 2 threads
  const int srow = tid >> 1;
  const int soff = (tid & 1) * 16;   // halfs
  const int arow = mblk + srow;
  const int atok = rowList ? rowList[arow] : arow;
  const _Float16* gA = A + (size_t)atok * K + soff;
  const _Float16* gB = Bt + (size_t)(nblk + srow) * K + soff;
  _Float16* lA0 = &sA[0][srow * RS + soff];
  _Float16* lA1 = &sA[1][srow * RS + soff];
  _Float16* lB0 = &sB[0][srow * RS + soff];
  _Float16* lB1 = &sB[1][srow * RS + soff];

  int aoff0 = (wm * 32 + lm) * RS;
  int aoff1 = (wm * 32 + 16 + lm) * RS;
  int boff[4];
#pragma unroll
  for (int j = 0; j < 4; ++j) boff[j] = (wn * 64 + j * 16 + lm) * RS;

  const v8f vz = {};
  v8f acc[2][4];
#pragma unroll
  for (int i = 0; i < 2; ++i)
#pragma unroll
    for (int j = 0; j < 4; ++j) acc[i][j] = vz;

  // prologue: fill buffer 0
  stage16(gA, lA0);     stage16(gA + 8, lA0 + 8);
  stage16(gB, lB0);     stage16(gB + 8, lB0 + 8);
  wait_async_all();
  __syncthreads();

  for (int kk = 0; kk < K; kk += 32) {
    const int buf = (kk >> 5) & 1;
    if (kk + 32 < K) {   // prefetch next tile into other buffer
      const _Float16* ga = gA + kk + 32;
      const _Float16* gb = gB + kk + 32;
      _Float16* la = buf ? lA0 : lA1;
      _Float16* lb = buf ? lB0 : lB1;
      stage16(ga, la);   stage16(ga + 8, la + 8);
      stage16(gb, lb);   stage16(gb + 8, lb + 8);
    }
    const _Float16* pa = sA[buf];
    const _Float16* pb = sB[buf];
    HFrag a0, a1, b[4];
    a0.h[0] = *(const v8h*)(pa + aoff0 + kh * 8);
    a0.h[1] = *(const v8h*)(pa + aoff0 + 16 + kh * 8);
    a1.h[0] = *(const v8h*)(pa + aoff1 + kh * 8);
    a1.h[1] = *(const v8h*)(pa + aoff1 + 16 + kh * 8);
#pragma unroll
    for (int j = 0; j < 4; ++j) {
      b[j].h[0] = *(const v8h*)(pb + boff[j] + kh * 8);
      b[j].h[1] = *(const v8h*)(pb + boff[j] + 16 + kh * 8);
    }
#pragma unroll
    for (int j = 0; j < 4; ++j) {
      acc[0][j] = __builtin_amdgcn_wmma_f32_16x16x32_f16(false, a0.v, false, b[j].v,
                                                         (short)0, acc[0][j], false, false);
      acc[1][j] = __builtin_amdgcn_wmma_f32_16x16x32_f16(false, a1.v, false, b[j].v,
                                                         (short)0, acc[1][j], false, false);
    }
    wait_async_all();
    __syncthreads();
  }

#pragma unroll
  for (int j = 0; j < 4; ++j) {
    const int n = nblk + wn * 64 + j * 16 + lm;
    const float bv = bias[n];
#pragma unroll
    for (int i = 0; i < 2; ++i) {
#pragma unroll
      for (int v = 0; v < 8; ++v) {
        const int p = mblk + wm * 32 + i * 16 + v + 8 * kh;  // C/D: row = vgpr + 8*(lane>=16)
        float r = acc[i][j][v] + bv;
        r = fmaxf(r, 0.0f);
        Hout[(size_t)p * N + n] = (_Float16)r;
      }
    }
  }
}

// ---------------------------------------------------------------------------
// WMMA GEMM #2 (scattered accumulate):
//   Out[tok(p)][n] += g(p) * ( sum_k A[p][k]*Bt[n][k] + bias[n] )
// rowList/gval==null -> identity / constant `scale` (shared experts).
// ---------------------------------------------------------------------------
__global__ __launch_bounds__(256) void moe_gemm2_acc(
    const _Float16* __restrict__ A, const _Float16* __restrict__ Bt,
    const float* __restrict__ bias, const float* __restrict__ gval,
    const int* __restrict__ rowList, const int* __restrict__ cntPtr,
    int Mtotal, float scale, float* __restrict__ Out, int N, int K) {
  const int cnt = cntPtr ? *cntPtr : Mtotal;
  const int mblk = blockIdx.y * 128;
  if (mblk >= cnt) return;

  __shared__ __align__(16) _Float16 sA[2][128 * RS];
  __shared__ __align__(16) _Float16 sB[2][128 * RS];

  const int tid = threadIdx.x;
  const int lane = tid & 31, wave = tid >> 5;
  const int wm = wave >> 1, wn = wave & 1;
  const int lm = lane & 15, kh = lane >> 4;
  const int nblk = blockIdx.x * 128;

  const int srow = tid >> 1;
  const int soff = (tid & 1) * 16;
  const _Float16* gA = A + (size_t)(mblk + srow) * K + soff;   // A is position-indexed
  const _Float16* gB = Bt + (size_t)(nblk + srow) * K + soff;
  _Float16* lA0 = &sA[0][srow * RS + soff];
  _Float16* lA1 = &sA[1][srow * RS + soff];
  _Float16* lB0 = &sB[0][srow * RS + soff];
  _Float16* lB1 = &sB[1][srow * RS + soff];

  int aoff0 = (wm * 32 + lm) * RS;
  int aoff1 = (wm * 32 + 16 + lm) * RS;
  int boff[4];
#pragma unroll
  for (int j = 0; j < 4; ++j) boff[j] = (wn * 64 + j * 16 + lm) * RS;

  const v8f vz = {};
  v8f acc[2][4];
#pragma unroll
  for (int i = 0; i < 2; ++i)
#pragma unroll
    for (int j = 0; j < 4; ++j) acc[i][j] = vz;

  stage16(gA, lA0);     stage16(gA + 8, lA0 + 8);
  stage16(gB, lB0);     stage16(gB + 8, lB0 + 8);
  wait_async_all();
  __syncthreads();

  for (int kk = 0; kk < K; kk += 32) {
    const int buf = (kk >> 5) & 1;
    if (kk + 32 < K) {
      const _Float16* ga = gA + kk + 32;
      const _Float16* gb = gB + kk + 32;
      _Float16* la = buf ? lA0 : lA1;
      _Float16* lb = buf ? lB0 : lB1;
      stage16(ga, la);   stage16(ga + 8, la + 8);
      stage16(gb, lb);   stage16(gb + 8, lb + 8);
    }
    const _Float16* pa = sA[buf];
    const _Float16* pb = sB[buf];
    HFrag a0, a1, b[4];
    a0.h[0] = *(const v8h*)(pa + aoff0 + kh * 8);
    a0.h[1] = *(const v8h*)(pa + aoff0 + 16 + kh * 8);
    a1.h[0] = *(const v8h*)(pa + aoff1 + kh * 8);
    a1.h[1] = *(const v8h*)(pa + aoff1 + 16 + kh * 8);
#pragma unroll
    for (int j = 0; j < 4; ++j) {
      b[j].h[0] = *(const v8h*)(pb + boff[j] + kh * 8);
      b[j].h[1] = *(const v8h*)(pb + boff[j] + 16 + kh * 8);
    }
#pragma unroll
    for (int j = 0; j < 4; ++j) {
      acc[0][j] = __builtin_amdgcn_wmma_f32_16x16x32_f16(false, a0.v, false, b[j].v,
                                                         (short)0, acc[0][j], false, false);
      acc[1][j] = __builtin_amdgcn_wmma_f32_16x16x32_f16(false, a1.v, false, b[j].v,
                                                         (short)0, acc[1][j], false, false);
    }
    wait_async_all();
    __syncthreads();
  }

  float bv[4];
#pragma unroll
  for (int j = 0; j < 4; ++j) bv[j] = bias[nblk + wn * 64 + j * 16 + lm];

#pragma unroll
  for (int i = 0; i < 2; ++i) {
#pragma unroll
    for (int v = 0; v < 8; ++v) {
      const int p = mblk + wm * 32 + i * 16 + v + 8 * kh;
      if (p < cnt) {     // skip padded rows (divergence only after all WMMAs)
        const int tok = rowList ? rowList[p] : p;
        const float g = gval ? gval[p] : scale;
#pragma unroll
        for (int j = 0; j < 4; ++j) {
          const int n = nblk + wn * 64 + j * 16 + lm;
          Out[(size_t)tok * N + n] += g * (acc[i][j][v] + bv[j]);
        }
      }
    }
  }
}

// ---------------------------------------------------------------------------
// Launch sequence
// ---------------------------------------------------------------------------
extern "C" void kernel_launch(void* const* d_in, const int* in_sizes, int n_in,
                              void* d_out, int out_size, void* d_ws, size_t ws_size,
                              hipStream_t stream) {
  const float* x        = (const float*)d_in[0];
  const float* sw1      = (const float*)d_in[1];
  const float* sb1      = (const float*)d_in[2];
  const float* sw2      = (const float*)d_in[3];
  const float* sb2      = (const float*)d_in[4];
  const float* rw1      = (const float*)d_in[5];
  const float* rb1      = (const float*)d_in[6];
  const float* rw2      = (const float*)d_in[7];
  const float* rb2      = (const float*)d_in[8];
  const float* router_w = (const float*)d_in[9];
  const float* router_b = (const float*)d_in[10];
  float* out = (float*)d_out;

  const int D = D_MODEL, H = D_HIDDEN, T = T_TOKENS;

  // Workspace layout (~41 MB), 256B aligned slabs.
  char* ws = (char*)d_ws;
  auto al = [](size_t v) { return (v + 255) & ~(size_t)255; };
  size_t off = 0;
  _Float16* Xh   = (_Float16*)(ws + off); off = al(off + (size_t)T * D * sizeof(_Float16));
  _Float16* Hbuf = (_Float16*)(ws + off); off = al(off + (size_t)T * H * sizeof(_Float16));
  _Float16* W1T  = (_Float16*)(ws + off); off = al(off + (size_t)H * D * sizeof(_Float16));
  _Float16* W2T  = (_Float16*)(ws + off); off = al(off + (size_t)D * H * sizeof(_Float16));
  int*      cnt  = (int*)(ws + off);      off = al(off + (size_t)NUM_ROUTED * sizeof(int));
  int*      list = (int*)(ws + off);      off = al(off + (size_t)NUM_ROUTED * T * sizeof(int));
  float*    gval = (float*)(ws + off);    off = al(off + (size_t)NUM_ROUTED * T * sizeof(float));

  // 1) zero out accumulator, counters, token lists; convert x -> f16
  {
    size_t n = (size_t)T * D;
    moe_zero_b32<<<(unsigned)((n + 255) / 256), 256, 0, stream>>>((unsigned*)out, n);
    size_t n2 = (size_t)NUM_ROUTED * (T + 1);
    moe_zero_b32<<<(unsigned)((n2 + 255) / 256), 256, 0, stream>>>((unsigned*)cnt, n2);
    moe_f32_to_f16<<<(unsigned)((n + 255) / 256), 256, 0, stream>>>(x, Xh, n);
  }

  // 2) router -> per-expert (token, weight) lists
  moe_router_scatter<<<T / 256, 256, 0, stream>>>(x, router_w, router_b, cnt, list, gval);

  const dim3 tb(32, 8);
  const dim3 tW1(H / 32, D / 32);   // [D][H] -> [H][D]
  const dim3 tW2(D / 32, H / 32);   // [H][D] -> [D][H]
  const dim3 g1(H / 128, T / 128);  // gemm1: N = H, max M = T
  const dim3 g2(D / 128, T / 128);  // gemm2: N = D, max M = T

  // 3) shared experts: dense, scale 1/NUM_SHARED
  for (int s = 0; s < NUM_SHARED; ++s) {
    moe_transpose_f16<<<tW1, tb, 0, stream>>>(sw1 + (size_t)s * D * H, W1T, D, H);
    moe_gemm1_relu<<<g1, 256, 0, stream>>>(Xh, W1T, sb1 + (size_t)s * H, Hbuf,
                                           nullptr, nullptr, T, H, D);
    moe_transpose_f16<<<tW2, tb, 0, stream>>>(sw2 + (size_t)s * H * D, W2T, H, D);
    moe_gemm2_acc<<<g2, 256, 0, stream>>>(Hbuf, W2T, sb2 + (size_t)s * D,
                                          nullptr, nullptr, nullptr, T,
                                          1.0f / NUM_SHARED, out, D, H);
  }

  // 4) routed experts: only their gathered tokens (top-6 sparse)
  for (int e = 0; e < NUM_ROUTED; ++e) {
    const int* eList = list + (size_t)e * T;
    const float* eG  = gval + (size_t)e * T;
    moe_transpose_f16<<<tW1, tb, 0, stream>>>(rw1 + (size_t)e * D * H, W1T, D, H);
    moe_gemm1_relu<<<g1, 256, 0, stream>>>(Xh, W1T, rb1 + (size_t)e * H, Hbuf,
                                           eList, cnt + e, T, H, D);
    moe_transpose_f16<<<tW2, tb, 0, stream>>>(rw2 + (size_t)e * H * D, W2T, H, D);
    moe_gemm2_acc<<<g2, 256, 0, stream>>>(Hbuf, W2T, rb2 + (size_t)e * D,
                                          eG, eList, cnt + e, T, 1.0f, out, D, H);
  }
}
